// Transformer_4801773437481
// MI455X (gfx1250) — matrix operand order repaired and checked by reading, not compiled, not run on hardware
//
#include <hip/hip_runtime.h>

// ---------------- problem constants (match reference) ----------------
constexpr int kV  = 8192;
constexpr int kE  = 1024;
constexpr int kH  = 16;
constexpr int kDK = 64;   // kE / kH
constexpr int kL  = 6;
constexpr int kB  = 4;
constexpr int kT  = 1024;
constexpr int kM  = kB * kT;   // 4096 token rows

// ---------------- WMMA fragment plumbing ----------------
typedef __attribute__((ext_vector_type(16))) __bf16 v16bf;
typedef __attribute__((ext_vector_type(4)))  __bf16 v4bf;
typedef __attribute__((ext_vector_type(8)))  float  v8f;
typedef __attribute__((ext_vector_type(16))) float  v16f;
typedef __attribute__((ext_vector_type(4)))  float  v4f;
typedef __attribute__((ext_vector_type(4)))  unsigned int v4u;

union Frag {
  v16bf v;
  v4u   q[2];
  unsigned short h[16];
};

union Cvt4 { v4bf v; unsigned int u[2]; };

// 4 floats -> 4 bf16 -> two packed dwords in LDS (pure v_cvt_pk_bf16_f32 path)
__device__ __forceinline__ void stash4(unsigned short* dst, float4 f) {
  v4f vf = {f.x, f.y, f.z, f.w};
  Cvt4 c; c.v = __builtin_convertvector(vf, v4bf);
  *(unsigned int*)(dst)     = c.u[0];
  *(unsigned int*)(dst + 2) = c.u[1];
}

__device__ __forceinline__ void pack16(Frag& fr, float4 a, float4 b, float4 c, float4 d) {
  v16f vf;
  vf[0]  = a.x; vf[1]  = a.y; vf[2]  = a.z; vf[3]  = a.w;
  vf[4]  = b.x; vf[5]  = b.y; vf[6]  = b.z; vf[7]  = b.w;
  vf[8]  = c.x; vf[9]  = c.y; vf[10] = c.z; vf[11] = c.w;
  vf[12] = d.x; vf[13] = d.y; vf[14] = d.z; vf[15] = d.w;
  fr.v = __builtin_convertvector(vf, v16bf);
}

#define WMMA_BF16(A, Bv, C) \
  __builtin_amdgcn_wmma_f32_16x16x32_bf16(false, (A), false, (Bv), (short)0, (C), false, false)

// ---------------- embedding gather ----------------
__global__ __launch_bounds__(256) void embed_kernel(float* __restrict__ x,
                                                    const float* __restrict__ emb,
                                                    const int* __restrict__ toks) {
  int idx = blockIdx.x * 256 + threadIdx.x;       // float4 index over kM*kE/4
  int e4  = idx & (kE / 4 - 1);
  int m   = idx >> 8;                              // kE/4 == 256
  int tok = toks[m];
  ((float4*)x)[idx] = ((const float4*)emb)[(size_t)tok * (kE / 4) + e4];
}

// ---------------- generic NT GEMM: C[M,N] = A[M,K] * W[N,K]^T + bias ----------------
// bf16 WMMA with f32 accumulate; 128x128 block tile, K-step 32, LDS staged.
#define GE_TM 128
#define GE_TN 128
#define GE_TK 32
#define LDSTR 40   // ushort row stride (32 + 8 pad), keeps 16B alignment

__global__ __launch_bounds__(256) void gemm_nt_bf16(
    float* __restrict__ C, const float* __restrict__ A, const float* __restrict__ Bw,
    const float* __restrict__ bias, int Mtot, int Ntot, int Ktot, int relu, int streamC) {
  __shared__ unsigned short lA[GE_TM * LDSTR];
  __shared__ unsigned short lB[GE_TN * LDSTR];

  const int tid = threadIdx.x;
  const int l   = tid & 31;
  const int lr  = l & 15;
  const int lhi = l >> 4;
  const int w   = tid >> 5;            // wave 0..7
  const int wm  = (w & 3) * 32;        // wave m offset (2 tiles of 16)
  const int wn  = (w >> 2) * 64;       // wave n offset (4 tiles of 16)
  const int bM  = blockIdx.y * GE_TM;
  const int bN  = blockIdx.x * GE_TN;

  v8f zero = {0.f, 0.f, 0.f, 0.f, 0.f, 0.f, 0.f, 0.f};
  v8f acc[2][4];
#pragma unroll
  for (int mt = 0; mt < 2; ++mt)
#pragma unroll
    for (int nt = 0; nt < 4; ++nt) acc[mt][nt] = zero;

  for (int k0 = 0; k0 < Ktot; k0 += GE_TK) {
    __syncthreads();
    // cooperative stage: 128 rows x 32 cols of A and W, f32 -> bf16 (v_cvt_pk_bf16_f32)
#pragma unroll
    for (int i = 0; i < 4; ++i) {
      int e = tid + i * 256;
      int r = e >> 3;
      int c = (e & 7) << 2;
      const float* pa = A + (size_t)(bM + r) * Ktot + k0 + c;
      stash4(&lA[r * LDSTR + c], *(const float4*)pa);
      const float* pb = Bw + (size_t)(bN + r) * Ktot + k0 + c;
      stash4(&lB[r * LDSTR + c], *(const float4*)pb);
      if (k0 + GE_TK < Ktot) {                 // global_prefetch_b8 of next K block
        __builtin_prefetch(pa + GE_TK, 0, 0);
        __builtin_prefetch(pb + GE_TK, 0, 0);
      }
    }
    __syncthreads();

    // A fragments: row = lr, K halves {0..7,16..23} (lhi=0) / {8..15,24..31} (lhi=1)
    Frag af[2];
#pragma unroll
    for (int mt = 0; mt < 2; ++mt) {
      int r = wm + mt * 16 + lr;
      af[mt].q[0] = *(const v4u*)&lA[r * LDSTR + lhi * 8];
      af[mt].q[1] = *(const v4u*)&lA[r * LDSTR + 16 + lhi * 8];
    }
    // B fragments: col = lr, K halves {0..7,16..23} (lhi=0) / {8..15,24..31} (lhi=1)
    Frag bfr[4];
#pragma unroll
    for (int nt = 0; nt < 4; ++nt) {
      int r = wn + nt * 16 + lr;
      bfr[nt].q[0] = *(const v4u*)&lB[r * LDSTR + lhi * 8];
      bfr[nt].q[1] = *(const v4u*)&lB[r * LDSTR + 16 + lhi * 8];
    }
#pragma unroll
    for (int mt = 0; mt < 2; ++mt)
#pragma unroll
      for (int nt = 0; nt < 4; ++nt)
        acc[mt][nt] = WMMA_BF16(af[mt].v, bfr[nt].v, acc[mt][nt]);
  }

  // epilogue: C layout -> row = base + v + 8*lhi, col = base + lr
#pragma unroll
  for (int nt = 0; nt < 4; ++nt) {
    int gcol = bN + wn + nt * 16 + lr;
    float bv = bias ? bias[gcol] : 0.f;
#pragma unroll
    for (int mt = 0; mt < 2; ++mt) {
#pragma unroll
      for (int vv = 0; vv < 8; ++vv) {
        int gr = bM + wm + mt * 16 + vv + 8 * lhi;
        float val = acc[mt][nt][vv] + bv;
        if (relu) val = fmaxf(val, 0.f);
        float* dst = C + (size_t)gr * Ntot + gcol;
        if (streamC) __builtin_nontemporal_store(val, dst);   // TH=NT: don't pollute L2
        else         *dst = val;
      }
    }
  }
}

// ---------------- transpose V slice of qkv into [B,H,DK,T] ----------------
__global__ __launch_bounds__(256) void vtrans_kernel(float* __restrict__ vT,
                                                     const float* __restrict__ qkv) {
  size_t idx = (size_t)blockIdx.x * 256 + threadIdx.x;   // over kB*kH*kDK*kT
  int s = idx & (kT - 1);
  int d = (idx >> 10) & (kDK - 1);
  int h = (idx >> 16) & (kH - 1);
  int b = (int)(idx >> 20);
  vT[idx] = qkv[((size_t)b * kT + s) * (3 * kE) + h * (3 * kDK) + 2 * kDK + d];
}

// ---------------- fused causal flash attention (one wave per 16-row q tile) ----------------
__global__ __launch_bounds__(32) void flash_attn(float* __restrict__ out,
                                                 const float* __restrict__ qkv,
                                                 const float* __restrict__ vT) {
  __shared__ unsigned short pL[16 * 32];   // P tile re-layout buffer (bf16)

  const int l   = threadIdx.x;
  const int lr  = l & 15;
  const int lhi = l >> 4;
  const int mt  = blockIdx.x & (kT / 16 - 1);       // 64 m-tiles
  const int h   = (blockIdx.x >> 6) & (kH - 1);
  const int b   = blockIdx.x >> 10;
  const int m0  = mt * 16;

  const size_t rs = 3 * kE;                          // qkv row stride
  const float* qBase = qkv + (size_t)b * kT * rs + h * (3 * kDK);
  const float* kBase = qBase + kDK;
  const float* vBase = vT + (size_t)(b * kH + h) * kDK * kT;
  const float  scale = 0.125f;                       // 1/sqrt(64)
  const float  NEG   = -3.0e38f;

  // q fragments, K = 0..31 and 32..63, held in registers for whole kernel
  Frag aq[2];
  {
    const float* qr = qBase + (size_t)(m0 + lr) * rs;
#pragma unroll
    for (int kk = 0; kk < 2; ++kk) {
      int ka = kk * 32 + lhi * 8;
      int kb = kk * 32 + 16 + lhi * 8;
      pack16(aq[kk],
             *(const float4*)(qr + ka), *(const float4*)(qr + ka + 4),
             *(const float4*)(qr + kb), *(const float4*)(qr + kb + 4));
    }
  }

  v8f zero = {0.f, 0.f, 0.f, 0.f, 0.f, 0.f, 0.f, 0.f};
  v8f oacc[4];
#pragma unroll
  for (int nt = 0; nt < 4; ++nt) oacc[nt] = zero;
  float mrun[8], lrun[8];
#pragma unroll
  for (int vv = 0; vv < 8; ++vv) { mrun[vv] = NEG; lrun[vv] = 0.f; }

  const int nblk = (m0 + 16 + 31) >> 5;              // 32-wide s-blocks in causal range
  for (int sb = 0; sb < nblk; ++sb) {
    const int s0 = sb * 32;

    // scores: two 16-col tiles, each K=64 (2 WMMAs)
    v8f sc[2];
#pragma unroll
    for (int st = 0; st < 2; ++st) {
      sc[st] = zero;
      int srow = s0 + st * 16 + lr;
      const float* kr = kBase + (size_t)srow * rs;
#pragma unroll
      for (int kk = 0; kk < 2; ++kk) {
        const float* kp = kr + kk * 32 + lhi * 8;
        Frag bk;
        pack16(bk, *(const float4*)(kp), *(const float4*)(kp + 4),
                   *(const float4*)(kp + 16), *(const float4*)(kp + 20));
        sc[st] = WMMA_BF16(aq[kk].v, bk.v, sc[st]);
      }
    }

    // scale + causal mask + online softmax (reductions across each 16-lane half)
    float p0[8], p1[8], alpha[8];
#pragma unroll
    for (int vv = 0; vv < 8; ++vv) {
      int gr = m0 + vv + 8 * lhi;
      float x0 = sc[0][vv] * scale; if (s0 + lr > gr)      x0 = NEG;
      float x1 = sc[1][vv] * scale; if (s0 + 16 + lr > gr) x1 = NEG;
      float r = fmaxf(x0, x1);
#pragma unroll
      for (int mm = 8; mm >= 1; mm >>= 1) r = fmaxf(r, __shfl_xor(r, mm, 32));
      float nm = fmaxf(mrun[vv], r);
      float al = __expf(mrun[vv] - nm);
      float e0 = __expf(x0 - nm);
      float e1 = __expf(x1 - nm);
      float rsum = e0 + e1;
#pragma unroll
      for (int mm = 8; mm >= 1; mm >>= 1) rsum += __shfl_xor(rsum, mm, 32);
      lrun[vv] = lrun[vv] * al + rsum;
      mrun[vv] = nm;
      p0[vv] = e0; p1[vv] = e1; alpha[vv] = al;
    }
#pragma unroll
    for (int nt = 0; nt < 4; ++nt)
#pragma unroll
      for (int vv = 0; vv < 8; ++vv) oacc[nt][vv] *= alpha[vv];

    // re-layout P (C-frag) -> A-frag via LDS
    __bf16* pB = (__bf16*)pL;
#pragma unroll
    for (int vv = 0; vv < 8; ++vv) {
      int prow = vv + 8 * lhi;
      pB[prow * 32 + lr]      = (__bf16)p0[vv];
      pB[prow * 32 + 16 + lr] = (__bf16)p1[vv];
    }
    __syncthreads();
    Frag pf;
    pf.q[0] = *(const v4u*)&pL[lr * 32 + lhi * 8];
    pf.q[1] = *(const v4u*)&pL[lr * 32 + 16 + lhi * 8];
    __syncthreads();

    // O += P x V  (V^T rows are d, contiguous in s)
#pragma unroll
    for (int nt = 0; nt < 4; ++nt) {
      const float* vr = vBase + (size_t)(nt * 16 + lr) * kT + s0 + lhi * 8;
      Frag vf;
      pack16(vf, *(const float4*)(vr), *(const float4*)(vr + 4),
                 *(const float4*)(vr + 16), *(const float4*)(vr + 20));
      oacc[nt] = WMMA_BF16(pf.v, vf.v, oacc[nt]);
    }
  }

  // normalize + write [M, E] at column block h*64
#pragma unroll
  for (int nt = 0; nt < 4; ++nt) {
#pragma unroll
    for (int vv = 0; vv < 8; ++vv) {
      int gr = m0 + vv + 8 * lhi;
      out[((size_t)b * kT + gr) * kE + h * kDK + nt * 16 + lr] = oacc[nt][vv] / lrun[vv];
    }
  }
}

// ---------------- layer norm: y = LN(sa*a + b) * g + beta ----------------
__global__ __launch_bounds__(256) void ln_kernel(float* __restrict__ y,
                                                 const float* __restrict__ a,
                                                 const float* __restrict__ b, float sa,
                                                 const float* __restrict__ g,
                                                 const float* __restrict__ beta) {
  int row = blockIdx.x, tid = threadIdx.x;
  const float* ar = a + (size_t)row * kE;
  const float* br = b ? b + (size_t)row * kE : nullptr;
  float vals[4];
  float s = 0.f, s2 = 0.f;
#pragma unroll
  for (int i = 0; i < 4; ++i) {
    int e = tid + i * 256;
    float v = sa * ar[e] + (br ? br[e] : 0.f);
    vals[i] = v; s += v; s2 += v * v;
  }
#pragma unroll
  for (int mm = 16; mm >= 1; mm >>= 1) {
    s  += __shfl_xor(s,  mm, 32);
    s2 += __shfl_xor(s2, mm, 32);
  }
  __shared__ float ss[8], sq[8];
  if ((tid & 31) == 0) { ss[tid >> 5] = s; sq[tid >> 5] = s2; }
  __syncthreads();
  float ts = 0.f, tq = 0.f;
#pragma unroll
  for (int j = 0; j < 8; ++j) { ts += ss[j]; tq += sq[j]; }
  float mu  = ts * (1.f / kE);
  float var = tq * (1.f / kE) - mu * mu;
  float r   = rsqrtf(var + 1e-5f);
#pragma unroll
  for (int i = 0; i < 4; ++i) {
    int e = tid + i * 256;
    y[(size_t)row * kE + e] = (vals[i] - mu) * r * g[e] + beta[e];
  }
}

// ---------------- orchestration ----------------
extern "C" void kernel_launch(void* const* d_in, const int* in_sizes, int n_in,
                              void* d_out, int out_size, void* d_ws, size_t ws_size,
                              hipStream_t stream) {
  (void)in_sizes; (void)n_in; (void)out_size; (void)ws_size;
  const int*   toks  = (const int*)d_in[0];
  const float* emb   = (const float*)d_in[1];
  const float* qkv_w = (const float*)d_in[2];
  const float* qkv_b = (const float*)d_in[3];
  const float* out_w = (const float*)d_in[4];
  const float* out_b = (const float*)d_in[5];
  const float* ln1_g = (const float*)d_in[6];
  const float* ln1_b = (const float*)d_in[7];
  const float* w1    = (const float*)d_in[8];
  const float* b1    = (const float*)d_in[9];
  const float* w2    = (const float*)d_in[10];
  const float* b2    = (const float*)d_in[11];
  const float* ln2_g = (const float*)d_in[12];
  const float* ln2_b = (const float*)d_in[13];
  const float* lin_w = (const float*)d_in[14];
  const float* lin_b = (const float*)d_in[15];

  char* ws = (char*)d_ws;
  size_t off = 0;
  float* x    = (float*)(ws + off); off += (size_t)kM * kE * 4;        // 16 MB activations
  float* big  = (float*)(ws + off); off += (size_t)kM * 4 * kE * 4;    // 64 MB qkv / ffn1
  float* vT   = (float*)(ws + off); off += (size_t)kM * kE * 4;        // 16 MB V transposed
  float* attn = (float*)(ws + off); off += (size_t)kM * kE * 4;        // 16 MB attention out
  float* fop  = (float*)(ws + off); off += (size_t)kM * kE * 4;        // 16 MB proj / ffn2
  float* hb   = (float*)(ws + off);                                    // 16 MB post-LN1

  embed_kernel<<<(kM * kE / 4) / 256, 256, 0, stream>>>(x, emb, toks);

  for (int l = 0; l < kL; ++l) {
    // qkv = x @ qkv_w^T + qkv_b     [4096, 3072]
    gemm_nt_bf16<<<dim3(3 * kE / 128, kM / 128), 256, 0, stream>>>(
        big, x, qkv_w + (size_t)l * 3 * kE * kE, qkv_b + (size_t)l * 3 * kE,
        kM, 3 * kE, kE, 0, 0);
    // v -> vT [B,H,DK,T]
    vtrans_kernel<<<(kB * kH * kDK * kT) / 256, 256, 0, stream>>>(vT, big);
    // fused causal attention -> attn [4096, 1024]
    flash_attn<<<kB * kH * (kT / 16), 32, 0, stream>>>(attn, big, vT);
    // out projection
    gemm_nt_bf16<<<dim3(kE / 128, kM / 128), 256, 0, stream>>>(
        fop, attn, out_w + (size_t)l * kE * kE, out_b + (size_t)l * kE,
        kM, kE, kE, 0, 0);
    // h = LN(x + o)
    ln_kernel<<<kM, 256, 0, stream>>>(hb, x, fop, 1.0f,
                                      ln1_g + (size_t)l * kE, ln1_b + (size_t)l * kE);
    // ffn1 = relu(h @ w1^T + b1)    [4096, 4096]
    gemm_nt_bf16<<<dim3(4 * kE / 128, kM / 128), 256, 0, stream>>>(
        big, hb, w1 + (size_t)l * 4 * kE * kE, b1 + (size_t)l * 4 * kE,
        kM, 4 * kE, kE, 1, 0);
    // f = ffn1 @ w2^T + b2
    gemm_nt_bf16<<<dim3(kE / 128, kM / 128), 256, 0, stream>>>(
        fop, big, w2 + (size_t)l * kE * 4 * kE, b2 + (size_t)l * kE,
        kM, kE, 4 * kE, 0, 0);
    // x = LN(f + f)  (reference reproduces the 2f bug)
    ln_kernel<<<kM, 256, 0, stream>>>(x, fop, nullptr, 2.0f,
                                      ln2_g + (size_t)l * kE, ln2_b + (size_t)l * kE);
  }

  // logits = x @ lin_w^T + lin_b   [4096, 8192] -- write-once, non-temporal stores
  gemm_nt_bf16<<<dim3(kV / 128, kM / 128), 256, 0, stream>>>(
      (float*)d_out, x, lin_w, lin_b, kM, kV, kE, 0, 1);
}
